// MOFTDoubleCrossAttnProcessor_74217034875002
// MI455X (gfx1250) — matrix operand order repaired and checked by reading, not compiled, not run on hardware
//
#include <hip/hip_runtime.h>
#include <hip/hip_bf16.h>

typedef __attribute__((ext_vector_type(16))) __bf16 v16bf;
typedef __attribute__((ext_vector_type(8)))  float  v8f;
typedef int async_v4i __attribute__((vector_size(16)));   // builtin's pointee type

#define DMODEL 1024
#define SEQ    2048
#define BATCH  2
#define NROWS  (BATCH * SEQ)      // 4096
#define NHEAD  16
#define DH     64

#if __has_builtin(__builtin_amdgcn_global_load_async_to_lds_b128) && \
    __has_builtin(__builtin_amdgcn_s_wait_asynccnt)
#define USE_ASYNC_LDS 1
#else
#define USE_ASYNC_LDS 0
#endif

// ---------------------------------------------------------------------------
// Fragment loaders (wave32; A/B use this code's contiguous k order, not the ISA table order)
// A 16x32 bf16: lane L -> M=L%16; elem e -> K = (L/16)*16 + e   (contiguous)
// B 32x16 bf16: lane L -> N=L%16; elem e -> K = (L/16)*16 + e   (contiguous)
// C/D 16x16 f32: lane L elem r -> M = r + (L/16)*8, N = L%16
// ---------------------------------------------------------------------------
__device__ __forceinline__ v16bf load_afrag_bf(const __bf16* __restrict__ p) {
  // bf16 source: p = &row[kb + half*16]; reads p[0..15] (2x b128)
  union { float4 f; __bf16 h[8]; } u0, u1;
  u0.f = *(const float4*)(p);
  u1.f = *(const float4*)(p + 8);
  v16bf r;
#pragma unroll
  for (int e = 0; e < 8; ++e) { r[e] = u0.h[e]; r[8 + e] = u1.h[e]; }
  return r;
}

__device__ __forceinline__ v16bf load_bfrag_bf(const __bf16* __restrict__ p) {
  // bf16 source: p = &row[kb + half*16]; 16 contiguous bf16 (2x b128)
  union { float4 f; __bf16 h[8]; } u0, u1;
  u0.f = *(const float4*)(p);
  u1.f = *(const float4*)(p + 8);
  v16bf r;
#pragma unroll
  for (int e = 0; e < 8; ++e) { r[e] = u0.h[e]; r[8 + e] = u1.h[e]; }
  return r;
}

__device__ __forceinline__ v16bf lds_bfrag_contig(const __bf16* __restrict__ p) {
  v16bf r;
#pragma unroll
  for (int e = 0; e < 16; ++e) r[e] = p[e];
  return r;
}

__device__ __forceinline__ v16bf lds_afrag(const __bf16* __restrict__ p) {
  v16bf r;
#pragma unroll
  for (int e = 0; e < 8; ++e) r[e] = p[e];
#pragma unroll
  for (int e = 0; e < 8; ++e) r[8 + e] = p[8 + e];
  return r;
}

__device__ __forceinline__ v16bf lds_vfrag(const __bf16* __restrict__ buf, int kbase, int d) {
  v16bf r;
#pragma unroll
  for (int e = 0; e < 16; ++e) r[e] = buf[(kbase + e) * 64 + d];
  return r;
}

// ---------------------------------------------------------------------------
// Cayley: Q = (I - S)^-1 (I + S), S = 0.5 (W - W^T).  Gauss-Jordan in LDS.
// ---------------------------------------------------------------------------
__global__ void __launch_bounds__(256) cayley64_kernel(const float* __restrict__ W1,
                                                       float* __restrict__ Q1) {
  __shared__ float A[64][65];
  __shared__ float Bm[64][65];
  __shared__ float fcol[64];
  const int tid = threadIdx.x;
  const float* W = W1 + (size_t)blockIdx.x * 4096;
  float*       Q = Q1 + (size_t)blockIdx.x * 4096;
  for (int i = tid; i < 4096; i += 256) {
    int r = i >> 6, c = i & 63;
    float s  = 0.5f * (W[r * 64 + c] - W[c * 64 + r]);
    float id = (r == c) ? 1.0f : 0.0f;
    A[r][c]  = id - s;
    Bm[r][c] = id + s;
  }
  __syncthreads();
  for (int p = 0; p < 64; ++p) {
    float inv = 1.0f / A[p][p];
    __syncthreads();
    if (tid < 128) {
      int c = tid & 63;
      if (tid < 64) A[p][c] *= inv; else Bm[p][c] *= inv;
    } else if (tid < 192) {
      fcol[tid - 128] = A[tid - 128][p];
    }
    __syncthreads();
    for (int i = tid; i < 64 * 128; i += 256) {
      int r = i >> 7, c2 = i & 127;
      if (r != p) {
        float f = fcol[r];
        if (c2 < 64) A[r][c2]       -= f * A[p][c2];
        else         Bm[r][c2 - 64] -= f * Bm[p][c2 - 64];
      }
    }
    __syncthreads();
  }
  for (int i = tid; i < 4096; i += 256) Q[i] = Bm[i >> 6][i & 63];
}

__global__ void __launch_bounds__(64) cayley16_kernel(const float* __restrict__ W2,
                                                      float* __restrict__ Q2) {
  __shared__ float A[16][17];
  __shared__ float Bm[16][17];
  __shared__ float fcol[16];
  const int tid = threadIdx.x;
  const float* W = W2 + (size_t)blockIdx.x * 256;
  float*       Q = Q2 + (size_t)blockIdx.x * 256;
  for (int i = tid; i < 256; i += 64) {
    int r = i >> 4, c = i & 15;
    float s  = 0.5f * (W[r * 16 + c] - W[c * 16 + r]);
    float id = (r == c) ? 1.0f : 0.0f;
    A[r][c]  = id - s;
    Bm[r][c] = id + s;
  }
  __syncthreads();
  for (int p = 0; p < 16; ++p) {
    float inv = 1.0f / A[p][p];
    __syncthreads();
    if (tid < 32) {
      int c = tid & 15;
      if (tid < 16) A[p][c] *= inv; else Bm[p][c] *= inv;
    } else if (tid < 48) {
      fcol[tid - 32] = A[tid - 32][p];
    }
    __syncthreads();
    for (int i = tid; i < 16 * 32; i += 64) {
      int r = i >> 5, c2 = i & 31;
      if (r != p) {
        float f = fcol[r];
        if (c2 < 16) A[r][c2]       -= f * A[p][c2];
        else         Bm[r][c2 - 16] -= f * Bm[p][c2 - 16];
      }
    }
    __syncthreads();
  }
  for (int i = tid; i < 256; i += 64) Q[i] = Bm[i >> 4][i & 15];
}

// ---------------------------------------------------------------------------
// chain = moft(moft(x, a), b). x viewed (rows, 16, 64). fp32 in -> bf16 out.
// step1: y[k,j] = sum_i x[k,i] Q1[k][i][j]
// step2: z[l,j] = sum_k y[k,j] Q2[j][k][l]   (out flat index l*64+j)
// ---------------------------------------------------------------------------
#define CH_ROWS 4
__global__ void __launch_bounds__(256) chain_kernel(const float* __restrict__ X,
    const float* __restrict__ Q1a, const float* __restrict__ Q2a,
    const float* __restrict__ Q1b, const float* __restrict__ Q2b,
    __bf16* __restrict__ Y) {
  __shared__ float xb[CH_ROWS][DMODEL];
  __shared__ float yb[CH_ROWS][DMODEL];
  const int tid = threadIdx.x;
  const size_t row0 = (size_t)blockIdx.x * CH_ROWS;
  for (int i = tid; i < CH_ROWS * DMODEL; i += 256)
    xb[i / DMODEL][i % DMODEL] = X[(row0 + i / DMODEL) * DMODEL + (i % DMODEL)];
  __syncthreads();
#pragma unroll
  for (int layer = 0; layer < 2; ++layer) {
    const float* Q1 = layer ? Q1b : Q1a;
    const float* Q2 = layer ? Q2b : Q2a;
    for (int o = tid; o < CH_ROWS * DMODEL; o += 256) {
      int r = o / DMODEL, kj = o % DMODEL;
      int k = kj >> 6, j = kj & 63;
      const float* xrow = &xb[r][k << 6];
      const float* qc   = Q1 + k * 4096 + j;
      float acc = 0.f;
#pragma unroll 8
      for (int i2 = 0; i2 < 64; ++i2) acc += xrow[i2] * qc[i2 * 64];
      yb[r][kj] = acc;
    }
    __syncthreads();
    for (int o = tid; o < CH_ROWS * DMODEL; o += 256) {
      int r = o / DMODEL, lj = o % DMODEL;
      int l = lj >> 6, j = lj & 63;
      const float* q2 = Q2 + j * 256 + l;
      float acc = 0.f;
#pragma unroll
      for (int k2 = 0; k2 < 16; ++k2) acc += yb[r][(k2 << 6) + j] * q2[k2 * 16];
      xb[r][lj] = acc;
    }
    __syncthreads();
  }
  // packed bf16 store (2 per dword)
  for (int i = tid; i < CH_ROWS * (DMODEL / 2); i += 256) {
    int r = i >> 9, c = (i & 511) << 1;
    union { unsigned u; __bf16 h[2]; } pk;
    pk.h[0] = (__bf16)xb[r][c];
    pk.h[1] = (__bf16)xb[r][c + 1];
    *(unsigned*)(Y + (row0 + r) * DMODEL + c) = pk.u;
  }
}

// ---------------------------------------------------------------------------
// one-time fp32 -> bf16 weight conversion (packed)
// ---------------------------------------------------------------------------
__global__ void __launch_bounds__(256) cvtw_kernel(const float* __restrict__ X,
                                                   __bf16* __restrict__ Y) {
  size_t i = ((size_t)blockIdx.x * 256 + threadIdx.x) * 2;
  union { unsigned u; __bf16 h[2]; } pk;
  pk.h[0] = (__bf16)X[i];
  pk.h[1] = (__bf16)X[i + 1];
  *(unsigned*)(Y + i) = pk.u;
}

// ---------------------------------------------------------------------------
// C[m,n] = sA[n]*sB[n]*(sum_k A[m,k] W[n,k]) (+bias[n]).  Wave: 16x64 tile.
// A, W in bf16; output bf16 (Cb) or f32 (Cf), whichever is non-null.
// Software-pipelined and 2x unrolled with ping-pong fragment registers so the
// loop-carried rotation needs no v_mov copies.
// ---------------------------------------------------------------------------
__global__ void __launch_bounds__(256) gemm16_kernel(const __bf16* __restrict__ A,
    const __bf16* __restrict__ W, const float* __restrict__ sA,
    const float* __restrict__ sB, const float* __restrict__ bias,
    float* __restrict__ Cf, __bf16* __restrict__ Cb) {
  const int lane  = threadIdx.x & 31;
  const int wid   = blockIdx.x * 8 + (threadIdx.x >> 5);
  const int tileM = wid >> 4;             // 0..255
  const int tileN = wid & 15;             // 0..15  (64 cols each)
  const int half  = lane >> 4;
  const int idx   = lane & 15;
  const int m     = tileM * 16 + idx;
  const int nb    = tileN * 64;
  v8f acc[4] = {v8f{}, v8f{}, v8f{}, v8f{}};
  const __bf16* arow = A + (size_t)m * DMODEL;

  v16bf a0 = load_afrag_bf(arow + half * 16);
  v16bf b0[4];
#pragma unroll
  for (int j = 0; j < 4; ++j)
    b0[j] = load_bfrag_bf(W + (size_t)(nb + j * 16 + idx) * DMODEL + half * 16);

  for (int kb = 0; kb < DMODEL; kb += 64) {
    // stage 1: prefetch k+32 into set 1, compute with set 0
    const int k1 = kb + 32;
    v16bf a1 = load_afrag_bf(arow + k1 + half * 16);
    v16bf b1[4];
#pragma unroll
    for (int j = 0; j < 4; ++j)
      b1[j] = load_bfrag_bf(W + (size_t)(nb + j * 16 + idx) * DMODEL + k1 + half * 16);
#pragma unroll
    for (int j = 0; j < 4; ++j)
      acc[j] = __builtin_amdgcn_wmma_f32_16x16x32_bf16(false, a0, false, b0[j],
                                                       (short)0, acc[j], false, false);
    // stage 2: prefetch k+64 (dummy wrap on last iter) into set 0, compute set 1
    const int k2 = (kb + 64 < DMODEL) ? kb + 64 : 0;
    a0 = load_afrag_bf(arow + k2 + half * 16);
#pragma unroll
    for (int j = 0; j < 4; ++j)
      b0[j] = load_bfrag_bf(W + (size_t)(nb + j * 16 + idx) * DMODEL + k2 + half * 16);
#pragma unroll
    for (int j = 0; j < 4; ++j)
      acc[j] = __builtin_amdgcn_wmma_f32_16x16x32_bf16(false, a1, false, b1[j],
                                                       (short)0, acc[j], false, false);
  }

#pragma unroll
  for (int j = 0; j < 4; ++j) {
    int n = nb + j * 16 + idx;
    float sc = sA[n] * sB[n];
    float bz = bias ? bias[n] : 0.0f;
    if (Cb) {
#pragma unroll
      for (int r = 0; r < 8; ++r) {
        int row = tileM * 16 + r + half * 8;
        Cb[(size_t)row * DMODEL + n] = (__bf16)(acc[j][r] * sc + bz);
      }
    } else {
#pragma unroll
      for (int r = 0; r < 8; ++r) {
        int row = tileM * 16 + r + half * 8;
        Cf[(size_t)row * DMODEL + n] = acc[j][r] * sc + bz;
      }
    }
  }
}

// ---------------------------------------------------------------------------
// Flash attention. Block: 4 waves x 16 queries. K/V double-buffered in LDS;
// async global->LDS b128 copies for block i+1 overlap WMMA/softmax of block i
// (async ops complete in order: s_wait_asynccnt<=4 drains the older buffer's
// copies while the 4 just-issued stay in flight).
// ---------------------------------------------------------------------------
__global__ void __launch_bounds__(128) attention_kernel(const __bf16* __restrict__ Q,
    const __bf16* __restrict__ K, const __bf16* __restrict__ V,
    float* __restrict__ O) {
  __shared__ __bf16 kbuf[2][32][64];
  __shared__ __bf16 vbuf[2][32][64];
  __shared__ __bf16 pbuf[4][16][32];
  const int tid  = threadIdx.x;
  const int wave = tid >> 5;
  const int lane = tid & 31;
  const int half = lane >> 4;
  const int idx  = lane & 15;
  const int qblk = blockIdx.x & 31;        // 32 blocks of 64 queries
  const int bh   = blockIdx.x >> 5;        // 0..31
  const int b    = bh >> 4;
  const int h    = bh & 15;
  const size_t base = (size_t)b * SEQ * DMODEL + (size_t)h * DH;
  const int qrow = qblk * 64 + wave * 16;

  const __bf16* qp = Q + base + (size_t)(qrow + idx) * DMODEL;
  v16bf qa0 = load_afrag_bf(qp + 0  + half * 16);   // dh 0..31
  v16bf qa1 = load_afrag_bf(qp + 32 + half * 16);   // dh 32..63

  v8f o0 = {}, o1 = {}, o2 = {}, o3 = {};
  float mrun[8], lrun[8];
#pragma unroll
  for (int r = 0; r < 8; ++r) { mrun[r] = -1e30f; lrun[r] = 0.0f; }
  const float sm = 0.125f;   // 1/sqrt(64)

#if USE_ASYNC_LDS
  auto issue_copies = [&](int kpos, int buf) {
#pragma unroll
    for (int c = 0; c < 2; ++c) {
      int lin = (c * 128 + tid) * 8;           // bf16 units, 16B granules
      int key = lin >> 6, d = lin & 63;
      const __bf16* gk = K + base + (size_t)(kpos + key) * DMODEL + d;
      const __bf16* gv = V + base + (size_t)(kpos + key) * DMODEL + d;
      __builtin_amdgcn_global_load_async_to_lds_b128(
          (async_v4i*)gk, (async_v4i*)(&kbuf[buf][0][0] + lin), 0, 0);
      __builtin_amdgcn_global_load_async_to_lds_b128(
          (async_v4i*)gv, (async_v4i*)(&vbuf[buf][0][0] + lin), 0, 0);
    }
  };
  issue_copies(0, 0);
#endif

  int cur = 0;
  for (int kpos = 0; kpos < SEQ; kpos += 32, cur ^= 1) {
#if USE_ASYNC_LDS
    __syncthreads();                               // readers of buf[cur^1] done
    const int knext = (kpos + 32 < SEQ) ? kpos + 32 : kpos;
    issue_copies(knext, cur ^ 1);                  // prefetch next block
    __builtin_amdgcn_s_wait_asynccnt(4);           // buf[cur]'s copies landed
    __syncthreads();
#else
    __syncthreads();
    for (int i = tid; i < 32 * 64; i += 128) {
      int key = i >> 6, d = i & 63;
      kbuf[cur][key][d] = K[base + (size_t)(kpos + key) * DMODEL + d];
      vbuf[cur][key][d] = V[base + (size_t)(kpos + key) * DMODEL + d];
    }
    __syncthreads();
#endif
    const __bf16* kc = &kbuf[cur][0][0];
    const __bf16* vc = &vbuf[cur][0][0];

    // scores: two 16x16 tiles over 32 keys, dh split into two K=32 chunks
    v8f s0 = {}, s1 = {};
    v16bf b00 = lds_bfrag_contig(kc + idx * 64        + 0  + half * 16);
    v16bf b01 = lds_bfrag_contig(kc + idx * 64        + 32 + half * 16);
    v16bf b10 = lds_bfrag_contig(kc + (16 + idx) * 64 + 0  + half * 16);
    v16bf b11 = lds_bfrag_contig(kc + (16 + idx) * 64 + 32 + half * 16);
    s0 = __builtin_amdgcn_wmma_f32_16x16x32_bf16(false, qa0, false, b00, (short)0, s0, false, false);
    s0 = __builtin_amdgcn_wmma_f32_16x16x32_bf16(false, qa1, false, b01, (short)0, s0, false, false);
    s1 = __builtin_amdgcn_wmma_f32_16x16x32_bf16(false, qa0, false, b10, (short)0, s1, false, false);
    s1 = __builtin_amdgcn_wmma_f32_16x16x32_bf16(false, qa1, false, b11, (short)0, s1, false, false);

    // online softmax over the 32-key block (rows split across 16-lane halves)
#pragma unroll
    for (int r = 0; r < 8; ++r) {
      float v0 = s0[r] * sm, v1 = s1[r] * sm;
      float mx = fmaxf(v0, v1);
#pragma unroll
      for (int off = 1; off < 16; off <<= 1) mx = fmaxf(mx, __shfl_xor(mx, off, 32));
      float mnew  = fmaxf(mrun[r], mx);
      float alpha = __expf(mrun[r] - mnew);
      float p0 = __expf(v0 - mnew);
      float p1 = __expf(v1 - mnew);
      float sum = p0 + p1;
#pragma unroll
      for (int off = 1; off < 16; off <<= 1) sum += __shfl_xor(sum, off, 32);
      lrun[r] = lrun[r] * alpha + sum;
      mrun[r] = mnew;
      o0[r] *= alpha; o1[r] *= alpha; o2[r] *= alpha; o3[r] *= alpha;
      int mloc = r + half * 8;
      pbuf[wave][mloc][idx]      = (__bf16)p0;
      pbuf[wave][mloc][16 + idx] = (__bf16)p1;
    }
    __builtin_amdgcn_wave_barrier();   // same-wave LDS ops are in-order

    // P as A-fragment (M=16 queries, K=32 keys); V as B-fragments
    v16bf pa  = lds_afrag(&pbuf[wave][idx][half * 16]);
    v16bf vf0 = lds_vfrag(vc, half * 16, idx);
    v16bf vf1 = lds_vfrag(vc, half * 16, 16 + idx);
    v16bf vf2 = lds_vfrag(vc, half * 16, 32 + idx);
    v16bf vf3 = lds_vfrag(vc, half * 16, 48 + idx);
    o0 = __builtin_amdgcn_wmma_f32_16x16x32_bf16(false, pa, false, vf0, (short)0, o0, false, false);
    o1 = __builtin_amdgcn_wmma_f32_16x16x32_bf16(false, pa, false, vf1, (short)0, o1, false, false);
    o2 = __builtin_amdgcn_wmma_f32_16x16x32_bf16(false, pa, false, vf2, (short)0, o2, false, false);
    o3 = __builtin_amdgcn_wmma_f32_16x16x32_bf16(false, pa, false, vf3, (short)0, o3, false, false);
  }

#pragma unroll
  for (int r = 0; r < 8; ++r) {
    float invl = 1.0f / lrun[r];
    size_t ob = base + (size_t)(qrow + r + half * 8) * DMODEL;
    O[ob + 0  + idx] = o0[r] * invl;
    O[ob + 16 + idx] = o1[r] * invl;
    O[ob + 32 + idx] = o2[r] * invl;
    O[ob + 48 + idx] = o3[r] * invl;
  }
}

// ---------------------------------------------------------------------------
extern "C" void kernel_launch(void* const* d_in, const int* in_sizes, int n_in,
                              void* d_out, int out_size, void* d_ws, size_t ws_size,
                              hipStream_t stream) {
  (void)in_sizes; (void)n_in; (void)out_size; (void)ws_size;
  const float* hs = (const float*)d_in[0];
  const float* W1 = (const float*)d_in[1];
  const float* W2 = (const float*)d_in[2];
  const float* wq = (const float*)d_in[3];
  const float* wk = (const float*)d_in[4];
  const float* wv = (const float*)d_in[5];
  const float* wo = (const float*)d_in[6];
  const float* bo = (const float*)d_in[7];
  const float* sc = (const float*)d_in[8];
  float* out = (float*)d_out;

  char* ws = (char*)d_ws;
  float*  Q1    = (float*) (ws + 0);            // 2,097,152 B
  float*  Q2    = (float*) (ws + 2097152);      //   524,288 B
  __bf16* tmpb  = (__bf16*)(ws + 2621440);      // 8,388,608 B (4096x1024 bf16)
  __bf16* qb    = (__bf16*)(ws + 11010048);     // 8,388,608 B
  __bf16* kb    = (__bf16*)(ws + 19398656);     // 8,388,608 B
  __bf16* vb    = (__bf16*)(ws + 27787264);     // 8,388,608 B
  float*  ad    = (float*) (ws + 36175872);     // 16,777,216 B (attn out f32)
  __bf16* wqb   = (__bf16*)(ws + 52953088);     // 2,097,152 B each
  __bf16* wkb   = (__bf16*)(ws + 55050240);
  __bf16* wvb   = (__bf16*)(ws + 57147392);
  __bf16* wob   = (__bf16*)(ws + 59244544);     // ends 61,341,696 B

  cayley64_kernel<<<128, 256, 0, stream>>>(W1, Q1);
  cayley16_kernel<<<512,  64, 0, stream>>>(W2, Q2);
  cvtw_kernel<<<2048, 256, 0, stream>>>(wq, wqb);
  cvtw_kernel<<<2048, 256, 0, stream>>>(wk, wkb);
  cvtw_kernel<<<2048, 256, 0, stream>>>(wv, wvb);
  cvtw_kernel<<<2048, 256, 0, stream>>>(wo, wob);

  auto q1 = [&](int i) { return Q1 + (size_t)i * 16 * 64 * 64; };
  auto q2 = [&](int i) { return Q2 + (size_t)i * 64 * 256; };

  chain_kernel<<<NROWS / CH_ROWS, 256, 0, stream>>>(hs, q1(0), q2(0), q1(1), q2(1), tmpb);
  gemm16_kernel<<<512, 256, 0, stream>>>(tmpb, wqb, sc + 0 * DMODEL, sc + 1 * DMODEL, nullptr, nullptr, qb);
  chain_kernel<<<NROWS / CH_ROWS, 256, 0, stream>>>(hs, q1(2), q2(2), q1(3), q2(3), tmpb);
  gemm16_kernel<<<512, 256, 0, stream>>>(tmpb, wkb, sc + 2 * DMODEL, sc + 3 * DMODEL, nullptr, nullptr, kb);
  chain_kernel<<<NROWS / CH_ROWS, 256, 0, stream>>>(hs, q1(4), q2(4), q1(5), q2(5), tmpb);
  gemm16_kernel<<<512, 256, 0, stream>>>(tmpb, wvb, sc + 4 * DMODEL, sc + 5 * DMODEL, nullptr, nullptr, vb);

  attention_kernel<<<BATCH * NHEAD * (SEQ / 64), 128, 0, stream>>>(qb, kb, vb, ad);

  chain_kernel<<<NROWS / CH_ROWS, 256, 0, stream>>>(ad, q1(6), q2(6), q1(7), q2(7), tmpb);
  gemm16_kernel<<<512, 256, 0, stream>>>(tmpb, wob, sc + 6 * DMODEL, sc + 7 * DMODEL, bo, out, nullptr);
}
